// ManualTransformerEncoderBlock_36240934044083
// MI455X (gfx1250) — matrix operand order repaired and checked by reading, not compiled, not run on hardware
//
#include <hip/hip_runtime.h>
#include <cmath>

// MI455X (gfx1250) transformer encoder block.
// All matmuls through v_wmma_f32_16x16x32_f16 (f16 inputs, f32 accumulate).
// Flash attention (no materialized S x S scores). wave32 throughout.
// GEMM: 64x64 output per wave (16 WMMA / 16 b128-loads per K-step ~ 32 FLOP/B).

#define TOK   8192      // B*S
#define BATCH 4
#define SEQ   2048
#define DM    768
#define DFF   3072
#define NH    12
#define DK    64

typedef __attribute__((ext_vector_type(16))) _Float16 v16h;
typedef __attribute__((ext_vector_type(8)))  _Float16 v8h;
typedef __attribute__((ext_vector_type(8)))  float    v8f;

union U16 { v16h v; v8h h[2]; float d[8]; _Float16 e[16]; };
union HP  { float f; _Float16 h[2]; };

__device__ __forceinline__ v8f wmma16(v16h a, v16h b, v8f c) {
  // (neg_a, A, neg_b, B, c_mod, C, reuse_a, reuse_b)
  return __builtin_amdgcn_wmma_f32_16x16x32_f16(false, a, false, b, (short)0, c, false, false);
}

// A-fragment (16x32, row-major source): lane holds row m=lane&15.
// elements 0..15 <- K = kb + half*16 + 0..15 (contiguous).
// p must already point at row + kb + half*16.
__device__ __forceinline__ v16h ldfragA(const _Float16* p) {
  U16 u; u.h[0] = *(const v8h*)(p); u.h[1] = *(const v8h*)(p + 8); return u.v;
}
// B-fragment (32x16) from Bt stored N x K row-major: lane holds col n=lane&15.
// elements 0..15 <- K = kb + half*16 + 0..15 (contiguous).
// p must already point at rowN + kb + half*16.
__device__ __forceinline__ v16h ldfragB(const _Float16* p) {
  U16 u; u.h[0] = *(const v8h*)(p); u.h[1] = *(const v8h*)(p + 8); return u.v;
}

// ---------------- weight convert + transpose: Wt[n*K+k] = (f16)W[k*N+n] ----
__global__ __launch_bounds__(256) void cvt_t_kernel(const float* __restrict__ W,
                                                    _Float16* __restrict__ Wt,
                                                    int K, int N) {
  size_t i = (size_t)blockIdx.x * 256 + threadIdx.x;  // over N*K outputs
  int n = (int)(i / K), k = (int)(i % K);
  Wt[i] = (_Float16)W[(size_t)k * N + n];
}

// ---------------- pack QKV biases into one 2304-float vector --------------
__global__ __launch_bounds__(256) void pack_bias_kernel(const float* __restrict__ a,
                                                        const float* __restrict__ b,
                                                        const float* __restrict__ c,
                                                        float* __restrict__ o) {
  int i = blockIdx.x * 256 + threadIdx.x;  // 0..2303
  const float* s = (i < DM) ? a : ((i < 2 * DM) ? b : c);
  o[i] = s[i % DM];
}

// ---------------- layernorm over last dim 768, f32 in -> f16 out ----------
__global__ __launch_bounds__(256) void ln_kernel(const float* __restrict__ x,
                                                 const float* __restrict__ g,
                                                 const float* __restrict__ b,
                                                 _Float16* __restrict__ out) {
  __shared__ float red[256];
  const int tid = threadIdx.x;
  const float* xr = x + (size_t)blockIdx.x * DM;
  float v0 = xr[tid], v1 = xr[tid + 256], v2 = xr[tid + 512];
  red[tid] = v0 + v1 + v2;
  __syncthreads();
  for (int st = 128; st > 0; st >>= 1) { if (tid < st) red[tid] += red[tid + st]; __syncthreads(); }
  float mu = red[0] * (1.0f / DM);
  __syncthreads();
  float d0 = v0 - mu, d1 = v1 - mu, d2 = v2 - mu;
  red[tid] = d0 * d0 + d1 * d1 + d2 * d2;
  __syncthreads();
  for (int st = 128; st > 0; st >>= 1) { if (tid < st) red[tid] += red[tid + st]; __syncthreads(); }
  float rstd = rsqrtf(red[0] * (1.0f / DM) + 1e-5f);
  _Float16* orow = out + (size_t)blockIdx.x * DM;
  orow[tid]       = (_Float16)(d0 * rstd * g[tid]       + b[tid]);
  orow[tid + 256] = (_Float16)(d1 * rstd * g[tid + 256] + b[tid + 256]);
  orow[tid + 512] = (_Float16)(d2 * rstd * g[tid + 512] + b[tid + 512]);
}

// ---------------- generic WMMA GEMM: C(MxN) = A(MxK) * Bt(NxK)^T ----------
// Wave computes 64x64; workgroup (8 waves) = 128 rows x 256 cols.
enum { MODE_QKV = 0, MODE_RESID = 2, MODE_GELU = 3 };

__global__ __launch_bounds__(256) void gemm_f16(const _Float16* __restrict__ A,
                                                const _Float16* __restrict__ Bt,
                                                const float* __restrict__ bias,
                                                const float* __restrict__ resid,
                                                void* __restrict__ outp,
                                                int N, int K, int mode) {
  const int lane = threadIdx.x & 31;
  const int wave = threadIdx.x >> 5;
  const int lm = lane & 15, half = lane >> 4;
  const int row0 = blockIdx.y * 128 + (wave & 1) * 64;   // 2 waves over rows
  const int col0 = blockIdx.x * 256 + (wave >> 1) * 64;  // 4 waves over cols

  v8f acc[16] = {};  // [i<4 row-tiles][j<4 col-tiles]
  const _Float16* arow[4];
#pragma unroll
  for (int i = 0; i < 4; ++i)
    arow[i] = A + (size_t)(row0 + i * 16 + lm) * K + half * 16;

  for (int kb = 0; kb < K; kb += 32) {
    if (kb + 128 < K) {  // stream-ahead hint for A (first-touch comes from HBM)
      __builtin_prefetch((const void*)(arow[0] + kb + 128), 0, 0);
      __builtin_prefetch((const void*)(arow[2] + kb + 128), 0, 0);
    }
    v16h a[4];
#pragma unroll
    for (int i = 0; i < 4; ++i) a[i] = ldfragA(arow[i] + kb);
#pragma unroll
    for (int j = 0; j < 4; ++j) {
      v16h bf = ldfragB(Bt + (size_t)(col0 + j * 16 + lm) * K + kb + half * 16);
#pragma unroll
      for (int i = 0; i < 4; ++i)
        acc[i * 4 + j] = wmma16(a[i], bf, acc[i * 4 + j]);
    }
  }

#pragma unroll
  for (int i = 0; i < 4; ++i) {
#pragma unroll
    for (int j = 0; j < 4; ++j) {
      const int gc = col0 + j * 16 + lm;
      const float bv = bias[gc];
      v8f c = acc[i * 4 + j];
#pragma unroll
      for (int r = 0; r < 8; ++r) {
        const int gm = row0 + i * 16 + 8 * half + r;
        float v = c[r] + bv;
        if (mode == MODE_RESID) {
          ((float*)outp)[(size_t)gm * N + gc] = v + resid[(size_t)gm * N + gc];
        } else if (mode == MODE_GELU) {
          float ge = 0.5f * v * (1.0f + erff(v * 0.70710678118f));
          ((_Float16*)outp)[(size_t)gm * N + gc] = (_Float16)ge;
        } else {  // MODE_QKV: gc in [0,2304): sect 0=Q,1=K (b,h,s,d); 2=V^T (b,h,d,s)
          const int sect = gc / DM, c768 = gc % DM;
          const int bb = gm >> 11, ss = gm & (SEQ - 1);
          const int hh = c768 >> 6, dd = c768 & (DK - 1);
          _Float16* base = (_Float16*)outp + (size_t)sect * TOK * DM;
          if (sect < 2)
            base[(((size_t)bb * NH + hh) * SEQ + ss) * DK + dd] = (_Float16)v;
          else
            base[(((size_t)bb * NH + hh) * DK + dd) * SEQ + ss] = (_Float16)v;
        }
      }
    }
  }
}

// ---------------- flash attention: one wave = 16 queries x Dk=64 ----------
// scores computed transposed (A = K tile, B = Q^T) so softmax is lane-local
// per query column; half-lane partials merged with shfl_xor(...,16).
__global__ __launch_bounds__(256) void attn_kernel(const _Float16* __restrict__ Q,
                                                   const _Float16* __restrict__ Kc,
                                                   const _Float16* __restrict__ Vt,
                                                   const float* __restrict__ mask,
                                                   _Float16* __restrict__ concat) {
  const int lane = threadIdx.x & 31;
  const int wave = threadIdx.x >> 5;
  const int lm = lane & 15, half = lane >> 4;

  const int qblock = blockIdx.x * 8 + wave;     // 6144 total
  const int b  = qblock / (NH * (SEQ / 16));
  const int r1 = qblock - b * (NH * (SEQ / 16));
  const int h  = r1 / (SEQ / 16);
  const int qb = (r1 % (SEQ / 16)) * 16;

  const _Float16* Qh = Q  + (size_t)(b * NH + h) * SEQ * DK;
  const _Float16* Kh = Kc + (size_t)(b * NH + h) * SEQ * DK;
  const _Float16* Vh = Vt + (size_t)(b * NH + h) * DK * SEQ;

  // Q^T fragments (B operand), Dk=64 -> two K-chunks of 32
  v16h qf0 = ldfragB(Qh + (size_t)(qb + lm) * DK + 0  + half * 16);
  v16h qf1 = ldfragB(Qh + (size_t)(qb + lm) * DK + 32 + half * 16);

  v8f o[4] = {};                 // out^T tiles: d = dt*16 + r + 8*half, col q = lm
  float m = -3.0e38f, ssum = 0.0f;
  const float scale = 0.125f;    // 1/sqrt(64)
  const float* maskrow = mask + (size_t)(qb + lm) * SEQ;

  for (int t = 0; t < SEQ; t += 32) {
    // scores^T, two 16-key tiles
    v8f st[2];
#pragma unroll
    for (int tile = 0; tile < 2; ++tile) {
      v16h k0 = ldfragA(Kh + (size_t)(t + tile * 16 + lm) * DK + 0  + half * 16);
      v16h k1 = ldfragA(Kh + (size_t)(t + tile * 16 + lm) * DK + 32 + half * 16);
      v8f z = {};
      z = wmma16(k0, qf0, z);
      z = wmma16(k1, qf1, z);
      st[tile] = z;
    }

    // scale + mask, lane-local running max
    float w[16];
    float lmx = -3.0e38f;
#pragma unroll
    for (int tile = 0; tile < 2; ++tile) {
      const float* mr = maskrow + t + tile * 16 + 8 * half;
#pragma unroll
      for (int r = 0; r < 8; ++r) {
        float v = st[tile][r] * scale + mr[r];
        w[tile * 8 + r] = v;
        lmx = fmaxf(lmx, v);
      }
    }
    float pmx  = __shfl_xor(lmx, 16, 32);
    float mnew = fmaxf(m, fmaxf(lmx, pmx));
    float alpha = __expf(m - mnew);
    m = mnew;

    float lsum = 0.0f;
    HP d0[4], d1[4];
#pragma unroll
    for (int i = 0; i < 4; ++i) {
      float e0 = __expf(w[2 * i]     - mnew), e1 = __expf(w[2 * i + 1]     - mnew);
      float f0 = __expf(w[8 + 2 * i] - mnew), f1 = __expf(w[8 + 2 * i + 1] - mnew);
      lsum += e0 + e1 + f0 + f1;
      d0[i].h[0] = (_Float16)e0; d0[i].h[1] = (_Float16)e1;
      d1[i].h[0] = (_Float16)f0; d1[i].h[1] = (_Float16)f1;
    }
    ssum = ssum * alpha + lsum;

    // repack exp weights into PV B-fragment (k = t_local = half*16 + e):
    // half=0 lane needs t 0..15  = own st0 | partner st0
    // half=1 lane needs t 16..31 = partner st1 | own st1
    U16 wf;
#pragma unroll
    for (int i = 0; i < 4; ++i) {
      float q0 = __shfl_xor(d0[i].f, 16, 32);
      float q1 = __shfl_xor(d1[i].f, 16, 32);
      wf.d[i]     = half ? q1       : d0[i].f;
      wf.d[4 + i] = half ? d1[i].f  : q0;
    }

    // out^T(d,q) += V^T(d,t) * w^T(t,q)
#pragma unroll
    for (int dt = 0; dt < 4; ++dt) {
#pragma unroll
      for (int r = 0; r < 8; ++r) o[dt][r] *= alpha;
      v16h vf = ldfragA(Vh + (size_t)(dt * 16 + lm) * SEQ + t + half * 16);
      o[dt] = wmma16(vf, wf.v, o[dt]);
    }
  }

  float stot = ssum + __shfl_xor(ssum, 16, 32);
  float inv = 1.0f / stot;
  const size_t orow = ((size_t)b * SEQ + qb + lm) * DM + h * DK;
#pragma unroll
  for (int dt = 0; dt < 4; ++dt)
#pragma unroll
    for (int r = 0; r < 8; ++r)
      concat[orow + dt * 16 + 8 * half + r] = (_Float16)(o[dt][r] * inv);
}

// --------------------------------------------------------------------------
extern "C" void kernel_launch(void* const* d_in, const int* in_sizes, int n_in,
                              void* d_out, int out_size, void* d_ws, size_t ws_size,
                              hipStream_t stream) {
  (void)in_sizes; (void)n_in; (void)out_size; (void)ws_size;
  const float* x    = (const float*)d_in[0];
  const float* mask = (const float*)d_in[1];
  const float* Wq = (const float*)d_in[2];  const float* bq = (const float*)d_in[3];
  const float* Wk = (const float*)d_in[4];  const float* bk = (const float*)d_in[5];
  const float* Wv = (const float*)d_in[6];  const float* bv = (const float*)d_in[7];
  const float* Wo = (const float*)d_in[8];  const float* bo = (const float*)d_in[9];
  const float* W1 = (const float*)d_in[10]; const float* b1 = (const float*)d_in[11];
  const float* W2 = (const float*)d_in[12]; const float* b2 = (const float*)d_in[13];
  const float* g1 = (const float*)d_in[14]; const float* be1 = (const float*)d_in[15];
  const float* g2 = (const float*)d_in[16]; const float* be2 = (const float*)d_in[17];

  char* ws = (char*)d_ws;
  auto alloc = [&](size_t bytes) { char* p = ws; ws += (bytes + 255) & ~(size_t)255; return p; };
  _Float16* xn    = (_Float16*)alloc((size_t)TOK * DM * 2);
  _Float16* Wqkvt = (_Float16*)alloc((size_t)3 * DM * DM * 2);   // [2304 x 768]
  _Float16* Wot   = (_Float16*)alloc((size_t)DM * DM * 2);
  _Float16* W1t   = (_Float16*)alloc((size_t)DM * DFF * 2);
  _Float16* W2t   = (_Float16*)alloc((size_t)DFF * DM * 2);
  float*    bqkv  = (float*)   alloc((size_t)3 * DM * 4);
  _Float16* QKVb  = (_Float16*)alloc((size_t)3 * TOK * DM * 2);  // Q | K | V^T
  _Float16* conc  = (_Float16*)alloc((size_t)TOK * DM * 2);
  float*    x1    = (float*)   alloc((size_t)TOK * DM * 4);
  _Float16* x1n   = (_Float16*)alloc((size_t)TOK * DM * 2);
  _Float16* hbuf  = (_Float16*)alloc((size_t)TOK * DFF * 2);

  // weight convert + transpose (one-time, tiny vs. GEMM cost)
  cvt_t_kernel<<<(DM * DM) / 256, 256, 0, stream>>>(Wq, Wqkvt,                0 * DM * DM ? DM : DM, DM);
  cvt_t_kernel<<<(DM * DM) / 256, 256, 0, stream>>>(Wk, Wqkvt + 1 * DM * DM,  DM, DM);
  cvt_t_kernel<<<(DM * DM) / 256, 256, 0, stream>>>(Wv, Wqkvt + 2 * DM * DM,  DM, DM);
  cvt_t_kernel<<<(DM * DM) / 256, 256, 0, stream>>>(Wo, Wot, DM, DM);
  cvt_t_kernel<<<(DM * DFF) / 256, 256, 0, stream>>>(W1, W1t, DM, DFF);   // -> (DFF x DM)
  cvt_t_kernel<<<(DFF * DM) / 256, 256, 0, stream>>>(W2, W2t, DFF, DM);   // -> (DM x DFF)
  pack_bias_kernel<<<(3 * DM) / 256, 256, 0, stream>>>(bq, bk, bv, bqkv);

  // LN1
  ln_kernel<<<TOK, 256, 0, stream>>>(x, g1, be1, xn);

  // fused QKV projection: one pass over xn, N = 2304
  gemm_f16<<<dim3(3 * DM / 256, TOK / 128), 256, 0, stream>>>(
      xn, Wqkvt, bqkv, nullptr, QKVb, 3 * DM, DM, MODE_QKV);

  // flash attention -> concat (f16)
  attn_kernel<<<(BATCH * NH * (SEQ / 16)) / 8, 256, 0, stream>>>(
      QKVb, QKVb + (size_t)TOK * DM, QKVb + (size_t)2 * TOK * DM, mask, conc);

  // x1 = x + concat @ Wo + bo
  gemm_f16<<<dim3(DM / 256, TOK / 128), 256, 0, stream>>>(conc, Wot, bo, x, x1, DM, DM, MODE_RESID);

  // LN2
  ln_kernel<<<TOK, 256, 0, stream>>>(x1, g2, be2, x1n);

  // h = gelu(x1n @ W1 + b1)
  gemm_f16<<<dim3(DFF / 256, TOK / 128), 256, 0, stream>>>(x1n, W1t, b1, nullptr, hbuf, DFF, DM, MODE_GELU);

  // out = x1 + h @ W2 + b2
  gemm_f16<<<dim3(DM / 256, TOK / 128), 256, 0, stream>>>(hbuf, W2t, b2, x1, (float*)d_out, DM, DFF, MODE_RESID);
}
